// SWD_62689342652980
// MI455X (gfx1250) — matrix operand order repaired and checked by reading, not compiled, not run on hardware
//
#include <hip/hip_runtime.h>
#include <hip/hip_bf16.h>

// ---------------------------------------------------------------------------
// MI455X (gfx1250) implementation.
// 3x 4096^3 GEMMs dominate: run them on V_WMMA_F32_16X16X32_BF16 (f32 accum).
// fp32 inputs are converted to bf16 while staging tiles into LDS.
// Wave32: 256-thread blocks = 8 waves, 128x128 block tile, 32x64 wave tile.
// ---------------------------------------------------------------------------

typedef __bf16 v16bf __attribute__((ext_vector_type(16)));
typedef float  v8f   __attribute__((ext_vector_type(8)));

#define TM 128
#define TN 128
#define TK 32
#define LDST 40   // ushort stride for LDS tiles; 80 bytes -> 16B-aligned rows

union Frag { v16bf v; unsigned int u[8]; };

__device__ __forceinline__ unsigned short f2bf(float f) {
  unsigned int u = __float_as_uint(f);
  u += 0x7FFFu + ((u >> 16) & 1u);   // round-to-nearest-even
  return (unsigned short)(u >> 16);
}

// ---------------------------------------------------------------------------
// C[i,j] = sum_k A[i*lda+k] * B[j*ldb+k] + bias   (f32 inputs, bf16 output)
// bias_row==1: bias indexed by i (GEMM1), else by j (GEMM2).
// ---------------------------------------------------------------------------
__global__ __launch_bounds__(256)
void gemm_nt_f32_bf16(const float* __restrict__ A, const float* __restrict__ B,
                      const float* __restrict__ bias, int bias_row,
                      unsigned short* __restrict__ C,
                      int K, int lda, int ldb, int ldc) {
  __shared__ unsigned short As[TM * LDST];
  __shared__ unsigned short Bs[TN * LDST];

  const int tid  = threadIdx.x;
  const int lane = tid & 31, wid = tid >> 5;
  const int wm = wid & 3, wn = wid >> 2;          // 4 x 2 wave grid
  const int m_blk = blockIdx.y * TM, n_blk = blockIdx.x * TN;

  v8f zero = {0.f,0.f,0.f,0.f,0.f,0.f,0.f,0.f};
  v8f acc[2][4];
#pragma unroll
  for (int i = 0; i < 2; ++i)
#pragma unroll
    for (int j = 0; j < 4; ++j) acc[i][j] = zero;

  for (int k0 = 0; k0 < K; k0 += TK) {
    // stage A tile 128x32 f32 -> bf16 LDS (float4 loads, uint2 LDS stores)
#pragma unroll
    for (int it = 0; it < 4; ++it) {
      int s = tid + it * 256;
      int row = s >> 3, c4 = (s & 7) << 2;
      float4 f = *(const float4*)(A + (size_t)(m_blk + row) * lda + k0 + c4);
      unsigned short h[4] = { f2bf(f.x), f2bf(f.y), f2bf(f.z), f2bf(f.w) };
      *(uint2*)&As[row * LDST + c4] = *(const uint2*)h;
    }
#pragma unroll
    for (int it = 0; it < 4; ++it) {
      int s = tid + it * 256;
      int row = s >> 3, c4 = (s & 7) << 2;
      float4 f = *(const float4*)(B + (size_t)(n_blk + row) * ldb + k0 + c4);
      unsigned short h[4] = { f2bf(f.x), f2bf(f.y), f2bf(f.z), f2bf(f.w) };
      *(uint2*)&Bs[row * LDST + c4] = *(const uint2*)h;
    }
    __syncthreads();

    const int hf = lane >> 4, l15 = lane & 15;
    Frag a[2], b[4];
#pragma unroll
    for (int mt = 0; mt < 2; ++mt) {
      int m = wm * 32 + mt * 16 + l15;
#pragma unroll
      for (int p = 0; p < 8; ++p) {
        int k = (p >> 2) * 16 + hf * 8 + (p & 3) * 2;   // 16-bit A layout
        a[mt].u[p] = *(const unsigned int*)&As[m * LDST + k];
      }
    }
#pragma unroll
    for (int nt = 0; nt < 4; ++nt) {
      int n = wn * 64 + nt * 16 + l15;
#pragma unroll
      for (int p = 0; p < 8; ++p) {
        int k = (p >> 2) * 16 + hf * 8 + (p & 3) * 2;   // 16-bit B layout
        b[nt].u[p] = *(const unsigned int*)&Bs[n * LDST + k];
      }
    }
#pragma unroll
    for (int mt = 0; mt < 2; ++mt)
#pragma unroll
      for (int nt = 0; nt < 4; ++nt)
        acc[mt][nt] = __builtin_amdgcn_wmma_f32_16x16x32_bf16(
            false, a[mt].v, false, b[nt].v, (short)0, acc[mt][nt], false, false);
    __syncthreads();
  }

  const int hf = lane >> 4, l15 = lane & 15;
#pragma unroll
  for (int mt = 0; mt < 2; ++mt)
#pragma unroll
    for (int nt = 0; nt < 4; ++nt)
#pragma unroll
      for (int e = 0; e < 8; ++e) {
        int m = m_blk + wm * 32 + mt * 16 + e + 8 * hf;
        int n = n_blk + wn * 64 + nt * 16 + l15;
        float v = acc[mt][nt][e] + (bias_row ? bias[m] : bias[n]);
        C[(size_t)m * ldc + n] = f2bf(v);
      }
}

// ---------------------------------------------------------------------------
// C[m,j] = sum_k A[m*lda+k] * B[k*ldb+j]   (bf16 inputs, f32 output)
// ---------------------------------------------------------------------------
__global__ __launch_bounds__(256)
void gemm_nn_bf16_f32(const unsigned short* __restrict__ A,
                      const unsigned short* __restrict__ B,
                      float* __restrict__ C,
                      int K, int lda, int ldb, int ldc) {
  __shared__ unsigned short As[TM * LDST];
  __shared__ unsigned short Bs[TN * LDST];

  const int tid  = threadIdx.x;
  const int lane = tid & 31, wid = tid >> 5;
  const int wm = wid & 3, wn = wid >> 2;
  const int m_blk = blockIdx.y * TM, n_blk = blockIdx.x * TN;

  v8f zero = {0.f,0.f,0.f,0.f,0.f,0.f,0.f,0.f};
  v8f acc[2][4];
#pragma unroll
  for (int i = 0; i < 2; ++i)
#pragma unroll
    for (int j = 0; j < 4; ++j) acc[i][j] = zero;

  for (int k0 = 0; k0 < K; k0 += TK) {
    // A tile 128x32 bf16: direct b128 copies
#pragma unroll
    for (int it = 0; it < 2; ++it) {
      int s = tid + it * 256;
      int row = s >> 2, c8 = (s & 3) << 3;
      uint4 d = *(const uint4*)(A + (size_t)(m_blk + row) * lda + k0 + c8);
      *(uint4*)&As[row * LDST + c8] = d;
    }
    // B tile 32(k) x 128(j): coalesced b128 loads, transposed scatter -> Bs[j][k]
#pragma unroll
    for (int it = 0; it < 2; ++it) {
      int s = tid + it * 256;
      int k = s >> 4, jg = (s & 15) << 3;
      uint4 d = *(const uint4*)(B + (size_t)(k0 + k) * ldb + n_blk + jg);
      unsigned short t8[8];
      *(uint4*)t8 = d;
#pragma unroll
      for (int jj = 0; jj < 8; ++jj) Bs[(jg + jj) * LDST + k] = t8[jj];
    }
    __syncthreads();

    const int hf = lane >> 4, l15 = lane & 15;
    Frag a[2], b[4];
#pragma unroll
    for (int mt = 0; mt < 2; ++mt) {
      int m = wm * 32 + mt * 16 + l15;
#pragma unroll
      for (int p = 0; p < 8; ++p) {
        int k = (p >> 2) * 16 + hf * 8 + (p & 3) * 2;
        a[mt].u[p] = *(const unsigned int*)&As[m * LDST + k];
      }
    }
#pragma unroll
    for (int nt = 0; nt < 4; ++nt) {
      int n = wn * 64 + nt * 16 + l15;
#pragma unroll
      for (int p = 0; p < 8; ++p) {
        int k = (p >> 2) * 16 + hf * 8 + (p & 3) * 2;
        b[nt].u[p] = *(const unsigned int*)&Bs[n * LDST + k];
      }
    }
#pragma unroll
    for (int mt = 0; mt < 2; ++mt)
#pragma unroll
      for (int nt = 0; nt < 4; ++nt)
        acc[mt][nt] = __builtin_amdgcn_wmma_f32_16x16x32_bf16(
            false, a[mt].v, false, b[nt].v, (short)0, acc[mt][nt], false, false);
    __syncthreads();
  }

  const int hf = lane >> 4, l15 = lane & 15;
#pragma unroll
  for (int mt = 0; mt < 2; ++mt)
#pragma unroll
    for (int nt = 0; nt < 4; ++nt)
#pragma unroll
      for (int e = 0; e < 8; ++e) {
        int m = m_blk + wm * 32 + mt * 16 + e + 8 * hf;
        int n = n_blk + wn * 64 + nt * 16 + l15;
        C[(size_t)m * ldc + n] = acc[mt][nt][e];
      }
}

// ---------------------------------------------------------------------------
// out[c,t] = band(ti)[t,c] + g[c,t]; band = sliding sum of width 2r+1 along c
// with the reference's left shift (pad = T2 - used). i read from device.
// ---------------------------------------------------------------------------
#define HALO 24
__global__ __launch_bounds__(256)
void band_transpose_add(const float* __restrict__ ti, const float* __restrict__ g,
                        const int* __restrict__ ip, float* __restrict__ out,
                        int T1, int T2) {
  __shared__ float tile[32][2 * HALO + 32 + 1];   // [32][81], odd stride
  const int c0 = blockIdx.x * 32;
  const int t0 = blockIdx.y * 32;
  const int tx = threadIdx.x;   // 0..31
  const int ty = threadIdx.y;   // 0..7

  for (int tr = ty; tr < 32; tr += 8) {
    const float* src = ti + (size_t)(t0 + tr) * T2;
    for (int w = tx; w < 32 + 2 * HALO; w += 32) {
      int cc = c0 - HALO + w;
      tile[tr][w] = (cc >= 0 && cc < T2) ? src[cc] : 0.0f;
    }
  }
  __syncthreads();

  const int wind = 6 + ip[0];
  int r = wind / 2 - 1;
  if (r > HALO / 2) r = HALO / 2;     // defensive bound
  const int used = (T2 / wind) * wind;
  const int pad  = T2 - used;

  for (int cl = ty; cl < 32; cl += 8) {
    const int cg = c0 + cl;
    float s = 0.0f;
    const int center = cg - pad;
    if (center >= 0) {
      int wbase = cl - pad - r + HALO;
      if (wbase < 0) wbase = 0;
      for (int d = 0; d <= 2 * r; ++d) s += tile[tx][wbase + d];
    }
    size_t o = (size_t)cg * T1 + t0 + tx;
    out[o] = s + g[o];
  }
}

// ---------------------------------------------------------------------------
// softmax over axis 0 (rows) of out (T2 x T1), in place.
// 256 threads = 64 columns x 4 row-chunks, LDS combine.
// ---------------------------------------------------------------------------
__global__ __launch_bounds__(256)
void softmax_axis0(float* __restrict__ out, int T2, int T1) {
  __shared__ float red[4][64];
  const int tid = threadIdx.x;
  const int cl = tid & 63;
  const int chunk = tid >> 6;
  const int col = blockIdx.x * 64 + cl;
  const int rp = T2 >> 2;
  const int cb = chunk * rp, ce = cb + rp;

  float m = -3.402823466e+38f;
  for (int c = cb; c < ce; ++c) m = fmaxf(m, out[(size_t)c * T1 + col]);
  red[chunk][cl] = m;
  __syncthreads();
  m = fmaxf(fmaxf(red[0][cl], red[1][cl]), fmaxf(red[2][cl], red[3][cl]));
  __syncthreads();

  float s = 0.0f;
  for (int c = cb; c < ce; ++c) s += expf(out[(size_t)c * T1 + col] - m);
  red[chunk][cl] = s;
  __syncthreads();
  s = red[0][cl] + red[1][cl] + red[2][cl] + red[3][cl];
  const float inv = 1.0f / s;

  for (int c = cb; c < ce; ++c) {
    size_t o = (size_t)c * T1 + col;
    out[o] = expf(out[o] - m) * inv;
  }
}

// ---------------------------------------------------------------------------
extern "C" void kernel_launch(void* const* d_in, const int* in_sizes, int n_in,
                              void* d_out, int out_size, void* d_ws, size_t ws_size,
                              hipStream_t stream) {
  const float* x1 = (const float*)d_in[0];
  const float* x2 = (const float*)d_in[1];
  const float* w1 = (const float*)d_in[2];
  const float* b1 = (const float*)d_in[3];
  const float* w2 = (const float*)d_in[4];
  const float* b2 = (const float*)d_in[5];
  const float* ga = (const float*)d_in[6];
  const int*   ip = (const int*)d_in[7];

  const int T1 = in_sizes[3];            // b1 length
  const int N  = in_sizes[0] / T1;       // x1 = N*T1
  const int T2 = in_sizes[1] / N;        // x2 = N*T2

  // workspace: y1t (T1xN bf16) | y2 (NxT2 bf16) | ti (T1xT2 f32)
  unsigned short* y1t = (unsigned short*)d_ws;
  unsigned short* y2  = y1t + (size_t)T1 * N;
  float*          ti  = (float*)(y2 + (size_t)N * T2);

  dim3 blk(256);

  // GEMM1: y1t[t,n] = sum_k w1[t,k]*x1[n,k] + b1[t]
  dim3 g1(N / TN, T1 / TM);
  gemm_nt_f32_bf16<<<g1, blk, 0, stream>>>(w1, x1, b1, 1, y1t, T1, T1, T1, N);

  // GEMM2: y2[n,t] = sum_k x2[n,k]*w2[t,k] + b2[t]
  dim3 g2(T1 / TN, N / TM);
  gemm_nt_f32_bf16<<<g2, blk, 0, stream>>>(x2, w2, b2, 0, y2, T2, T2, T1, T1);

  // GEMM3: ti[m,j] = sum_n y1t[m,n]*y2[n,j]
  dim3 g3(T2 / TN, T1 / TM);
  gemm_nn_bf16_f32<<<g3, blk, 0, stream>>>(y1t, y2, ti, N, N, T1, T2);

  // band + transpose + global_att add
  dim3 b4(32, 8);
  dim3 g4(T2 / 32, T1 / 32);
  band_transpose_add<<<g4, b4, 0, stream>>>(ti, ga, ip, (float*)d_out, T1, T2);

  // softmax over axis 0
  softmax_axis0<<<T1 / 64, blk, 0, stream>>>((float*)d_out, T2, T1);
}
